// OptimalSubgraphGNN_35888746726116
// MI455X (gfx1250) — matrix operand order repaired and checked by reading, not compiled, not run on hardware
//
#include <hip/hip_runtime.h>
#include <stdint.h>

// ---- CDNA5 WMMA vector types (wave32) ----
typedef __attribute__((ext_vector_type(16))) __bf16          v16bf;
typedef __attribute__((ext_vector_type(16))) unsigned short  v16u;
typedef __attribute__((ext_vector_type(8)))  float           v8f;

#define NN 2048          // nodes
#define NTI 128          // 2048/16 row tiles
#define NKC 64           // 2048/32 K chunks

__device__ __forceinline__ unsigned short f2bf(float f) {
  unsigned int u = __float_as_uint(f);
  u += 0x7FFFu + ((u >> 16) & 1u);     // round-to-nearest-even
  return (unsigned short)(u >> 16);
}

// =====================================================================
// Kernel 1: ATpack[ti][kc][lane][e] = bf16(relu(adj[k][i]))  (A-operand layout of A^T)
//   lane: M = lane&15 (i offset), half = lane>>4
//   e   : k_local = (e&7) + 8*half + 16*(e>>3)
// =====================================================================
__global__ void k_prep_adj(const float* __restrict__ adj, unsigned short* __restrict__ atp) {
  int id   = blockIdx.x * blockDim.x + threadIdx.x;   // 128*64*32 = 262144
  int lane = id & 31;
  int kc   = (id >> 5) & 63;
  int ti   = id >> 11;
  int half = lane >> 4;
  int col  = ti * 16 + (lane & 15);
  v16u v;
#pragma unroll
  for (int e = 0; e < 16; ++e) {
    int kl = (e & 7) + 8 * half + 16 * (e >> 3);
    float a = adj[(size_t)(kc * 32 + kl) * NN + col];
    v[e] = f2bf(a > 0.f ? a : 0.f);
  }
  *(v16u*)(atp + (size_t)id * 16) = v;
}

// =====================================================================
// Kernel 2: P = H @ W  (small GEMM), emitted directly in B-operand layout
//   Ppack[kc][nt][lane][e]: lane = ((k_local/8)%2)*16 + n%16, e = k_local%8 + 8*(k_local/16)
// =====================================================================
__global__ void k_hw_pack(const float* __restrict__ H, const float* __restrict__ W,
                          int inC, int u, unsigned short* __restrict__ pp) {
  int id = blockIdx.x * blockDim.x + threadIdx.x;
  if (id >= NN * u) return;
  int n = id % u, k = id / u;
  float acc = 0.f;
  for (int c = 0; c < inC; ++c) acc += H[(size_t)k * inC + c] * W[(size_t)c * u + n];
  int kcid = k >> 5, kl = k & 31;
  int lane = (((kl >> 3) & 1) << 4) + (n & 15);
  int nt = n >> 4, e = (kl & 7) + 8 * (kl >> 4);
  int NTn = u >> 4;
  pp[((size_t)((kcid * NTn + nt) * 32 + lane)) * 16 + e] = f2bf(acc);
}

// =====================================================================
// Kernel 3: H_out = relu(A^T @ P + b) via v_wmma_f32_16x16x32_bf16
//   one wave per (row-tile ti, col-tile nt); 64-deep K loop
// =====================================================================
__global__ void k_gcn(const unsigned short* __restrict__ atp,
                      const unsigned short* __restrict__ pp,
                      const float* __restrict__ bias,
                      float* __restrict__ hout, int u) {
  int lane = threadIdx.x & 31;
  int nt   = threadIdx.x >> 5;
  int ti   = blockIdx.x;
  int NTn  = u >> 4;
  v8f acc = {};
  for (int kc = 0; kc < NKC; ++kc) {
    v16u av = *(const v16u*)(atp + ((size_t)((ti * NKC + kc) * 32 + lane)) * 16);
    v16u bv = *(const v16u*)(pp  + ((size_t)((kc * NTn + nt) * 32 + lane)) * 16);
    v16bf a = __builtin_bit_cast(v16bf, av);
    v16bf b = __builtin_bit_cast(v16bf, bv);
    acc = __builtin_amdgcn_wmma_f32_16x16x32_bf16(false, a, false, b, (short)0, acc, false, false);
  }
  int n = lane & 15, g = lane >> 4;
  float bb = bias[nt * 16 + n];
#pragma unroll
  for (int r = 0; r < 8; ++r) {
    float v = acc[r] + bb;
    v = v > 0.f ? v : 0.f;
    hout[(size_t)(ti * 16 + r + 8 * g) * u + nt * 16 + n] = v;   // D: vgpr r -> M=r+8*(lane/16), N=lane%16
  }
}

// =====================================================================
// Kernel 4: tu[i][c] : c<32 -> t'=h@Wa+mb0 ; c>=32 -> u=h@Wb  (Wa=mW0[0:48], Wb=mW0[48:96])
// =====================================================================
__global__ void k_tu(const float* __restrict__ h, const float* __restrict__ mW0,
                     const float* __restrict__ mb0, float* __restrict__ tu) {
  int id = blockIdx.x * blockDim.x + threadIdx.x;   // 2048*64
  int c = id & 63, i = id >> 6;
  int cc = c & 31;
  int rbase = (c < 32) ? 0 : 48;
  float acc = (c < 32) ? mb0[cc] : 0.f;
  for (int d = 0; d < 48; ++d) acc += h[(size_t)i * 48 + d] * mW0[(size_t)(rbase + d) * 32 + cc];
  tu[id] = acc;
}

// =====================================================================
// Kernel 5: fused edge MLP. Wave handles (i, 16 pairs j0..j0+15):
//   build s1 (16x32 bf16, A-layout) in registers, WMMA vs W1 (C = b1),
//   relu, 16->1 dot via shfl_xor reduction, sigmoid, store o.
// =====================================================================
union F4 { float4 v; float f[4]; };

__global__ void k_edge(const float* __restrict__ adj, const float* __restrict__ tu,
                       const float* __restrict__ mW0, const float* __restrict__ mW1,
                       const float* __restrict__ mb1, const float* __restrict__ mW2,
                       const float* __restrict__ mb2, float* __restrict__ obuf) {
  int i    = blockIdx.x;
  int lane = threadIdx.x & 31;
  int wid  = threadIdx.x >> 5;
  int m = lane & 15, g = lane >> 4;

  // This lane's 16 channels: k = 8g+e (e<8) and 16+8g+(e-8) (e>=8) -> two float4-pair runs
  float tK[16], wcK[16];
  {
    const float4* tp = (const float4*)(tu + (size_t)i * 64);           // t' row
    const float4* wp = (const float4*)(mW0 + 96 * 32);                 // wc row
    F4 a0, a1, a2, a3, b0, b1, b2, b3;
    a0.v = tp[2 * g];     a1.v = tp[2 * g + 1];
    a2.v = tp[4 + 2 * g]; a3.v = tp[5 + 2 * g];
    b0.v = wp[2 * g];     b1.v = wp[2 * g + 1];
    b2.v = wp[4 + 2 * g]; b3.v = wp[5 + 2 * g];
#pragma unroll
    for (int q = 0; q < 4; ++q) {
      tK[q] = a0.f[q];  tK[4 + q] = a1.f[q];  tK[8 + q] = a2.f[q];  tK[12 + q] = a3.f[q];
      wcK[q] = b0.f[q]; wcK[4 + q] = b1.f[q]; wcK[8 + q] = b2.f[q]; wcK[12 + q] = b3.f[q];
    }
  }
  // Resident B tile: W1 (32x16) in B-operand layout: lane n=m, e -> k = (e&7)+8g+16*(e>>3)
  v16u bu;
#pragma unroll
  for (int e = 0; e < 16; ++e) bu[e] = f2bf(mW1[(size_t)((e & 7) + 8 * g + 16 * (e >> 3)) * 16 + m]);
  v16bf bmat = __builtin_bit_cast(v16bf, bu);

  v8f cb;
  float b1v = mb1[m];
#pragma unroll
  for (int r = 0; r < 8; ++r) cb[r] = b1v;
  float w2  = mW2[m];
  float b2v = mb2[0];

  for (int jt = wid; jt < 128; jt += 8) {
    int j0 = jt * 16;
    int jm = j0 + m;
    float aij = adj[(size_t)i * NN + jm];

    float uK[16];
    {
      const float4* up = (const float4*)(tu + (size_t)jm * 64 + 32);
      F4 u0, u1, u2, u3;
      u0.v = up[2 * g];     u1.v = up[2 * g + 1];
      u2.v = up[4 + 2 * g]; u3.v = up[5 + 2 * g];
#pragma unroll
      for (int q = 0; q < 4; ++q) {
        uK[q] = u0.f[q]; uK[4 + q] = u1.f[q]; uK[8 + q] = u2.f[q]; uK[12 + q] = u3.f[q];
      }
    }
    v16u au;
#pragma unroll
    for (int e = 0; e < 16; ++e) {
      float s = tK[e] + uK[e] + aij * wcK[e];
      au[e] = f2bf(s > 0.f ? s : 0.f);
    }
    v16bf amat = __builtin_bit_cast(v16bf, au);
    v8f d = __builtin_amdgcn_wmma_f32_16x16x32_bf16(false, amat, false, bmat, (short)0, cb, false, false);

    float red[8];
#pragma unroll
    for (int r = 0; r < 8; ++r) {
      float v = d[r] > 0.f ? d[r] : 0.f;   // relu(s2)
      v *= w2;                             // * mW2[n]
      v += __shfl_xor(v, 1);
      v += __shfl_xor(v, 2);
      v += __shfl_xor(v, 4);
      v += __shfl_xor(v, 8);               // sum over n within 16-lane group
      red[r] = v;
    }
    if (m == 0) {                          // lane0 -> rows 0..7, lane16 -> rows 8..15
#pragma unroll
      for (int r = 0; r < 8; ++r) {
        int j = j0 + r + 8 * g;
        float av = adj[(size_t)i * NN + j];
        float s = red[r] + b2v + (av * 2.f - 1.f) * 3.f;
        obuf[(size_t)i * NN + j] = 1.f / (1.f + __expf(-s));
      }
    }
  }
}

// =====================================================================
// Kernel 6: out = 0.5*(o + o^T), zero diagonal
// =====================================================================
__global__ void k_sym(const float* __restrict__ o, float* __restrict__ out) {
  size_t id = (size_t)blockIdx.x * blockDim.x + threadIdx.x;   // 2048*2048
  int j = (int)(id & 2047);
  int i = (int)(id >> 11);
  out[id] = (i == j) ? 0.f : 0.5f * (o[(size_t)i * NN + j] + o[(size_t)j * NN + i]);
}

// =====================================================================
extern "C" void kernel_launch(void* const* d_in, const int* in_sizes, int n_in,
                              void* d_out, int out_size, void* d_ws, size_t ws_size,
                              hipStream_t stream) {
  const float* x   = (const float*)d_in[0];
  const float* adj = (const float*)d_in[1];
  const float* gW0 = (const float*)d_in[2];  const float* gb0 = (const float*)d_in[3];
  const float* gW1 = (const float*)d_in[4];  const float* gb1 = (const float*)d_in[5];
  const float* gW2 = (const float*)d_in[6];  const float* gb2 = (const float*)d_in[7];
  const float* mW0 = (const float*)d_in[8];  const float* mb0 = (const float*)d_in[9];
  const float* mW1 = (const float*)d_in[10]; const float* mb1 = (const float*)d_in[11];
  const float* mW2 = (const float*)d_in[12]; const float* mb2 = (const float*)d_in[13];

  char* ws = (char*)d_ws;
  unsigned short* atp = (unsigned short*)ws;                       // 8 MiB   (2048^2 bf16, A-layout)
  unsigned short* pp  = (unsigned short*)(ws + (8u << 20));        // 256 KiB (B-layout P)
  float* hA   = (float*)(ws + (9u << 20));                         // 512 KiB
  float* hB   = hA  + NN * 64;                                     // 512 KiB
  float* tub  = hB  + NN * 64;                                     // 512 KiB
  float* obuf = tub + NN * 64;                                     // 16 MiB

  // 1) pack relu(adj)^T into WMMA A-operand layout
  k_prep_adj<<<1024, 256, 0, stream>>>(adj, atp);

  // 2) three GCN layers: P = h@W (packed), then h = relu(A^T P + b) via WMMA
  k_hw_pack<<<(NN * 64 + 255) / 256, 256, 0, stream>>>(x, gW0, 64, 64, pp);
  k_gcn<<<NTI, 4 * 32, 0, stream>>>(atp, pp, gb0, hA, 64);
  k_hw_pack<<<(NN * 64 + 255) / 256, 256, 0, stream>>>(hA, gW1, 64, 64, pp);
  k_gcn<<<NTI, 4 * 32, 0, stream>>>(atp, pp, gb1, hB, 64);
  k_hw_pack<<<(NN * 48 + 255) / 256, 256, 0, stream>>>(hB, gW2, 64, 48, pp);
  k_gcn<<<NTI, 3 * 32, 0, stream>>>(atp, pp, gb2, hA, 48);

  // 3) t' = h@Wa + b0, u = h@Wb
  k_tu<<<(NN * 64) / 256, 256, 0, stream>>>(hA, mW0, mb0, tub);

  // 4) fused edge MLP (WMMA) -> o
  k_edge<<<NN, 256, 0, stream>>>(adj, tub, mW0, mW1, mb1, mW2, mb2, obuf);

  // 5) symmetrize + zero diagonal
  k_sym<<<(NN * NN) / 256, 256, 0, stream>>>(obuf, (float*)d_out);
}
